// Attention_26585847562539
// MI455X (gfx1250) — matrix operand order repaired and checked by reading, not compiled, not run on hardware
//
#include <hip/hip_runtime.h>
#include <hip/hip_bf16.h>

typedef __attribute__((ext_vector_type(16))) _Float16 v16h;
typedef __attribute__((ext_vector_type(8)))  _Float16 v8h;
typedef __attribute__((ext_vector_type(4)))  _Float16 v4h;
typedef __attribute__((ext_vector_type(4)))  float    v4f;
typedef __attribute__((ext_vector_type(8)))  float    v8f;

static constexpr int Bn = 4, Sn = 2048, Dn = 1024, Hn = 16, HDn = 64;

__device__ __forceinline__ v16h mk_frag(v8h lo, v8h hi) {
  v16h r;
#pragma unroll
  for (int i = 0; i < 8; ++i) { r[i] = lo[i]; r[i + 8] = hi[i]; }
  return r;
}

__device__ __forceinline__ v8f wmma16(v16h a, v16h b, v8f c) {
  // (neg_a, A, neg_b, B, c_mod, C, reuse_a, reuse_b)
  return __builtin_amdgcn_wmma_f32_16x16x32_f16(false, a, false, b, (short)0, c, false, false);
}

// A fragment (16x32 f16, row-major source): lane half h holds K = {h*8..h*8+7, h*8+16..h*8+23}
__device__ __forceinline__ v16h load_a(const _Float16* rowp, int half) {
  const int ko = half * 8;
  v8h lo = *(const v8h*)(rowp + ko);
  v8h hi = *(const v8h*)(rowp + ko + 16);
  return mk_frag(lo, hi);
}

// B fragment (32x16 f16, per-column contiguous K source): lane half h holds K = {h*8..h*8+7, h*8+16..h*8+23}
__device__ __forceinline__ v16h load_b(const _Float16* colp, int half) {
  const int ko = half * 8;
  v8h lo = *(const v8h*)(colp + ko);
  v8h hi = *(const v8h*)(colp + ko + 16);
  return mk_frag(lo, hi);
}

__device__ __forceinline__ v4h cvt4(v4f t) {
  v4h p;
#pragma unroll
  for (int e = 0; e < 4; ++e) p[e] = (_Float16)t[e];
  return p;
}

// ---------------- Kernel 1: QKV projection  x[8192,1024] @ w[1024,3072] ----------------
__global__ __launch_bounds__(256) void qkv_gemm_kernel(
    const float* __restrict__ x, const float* __restrict__ w,
    const float* __restrict__ bias, _Float16* __restrict__ qh,
    float* __restrict__ present)
{
  constexpr int K = Dn, N = 3 * Dn;
  __shared__ _Float16 As[128][40];   // [m][k]
  __shared__ _Float16 Bs[128][40];   // [n][k] (transposed)
  const int tid = threadIdx.x;
  const int wv = tid >> 5, lane = tid & 31, half = lane >> 4, lr = lane & 15;
  const int wm = wv & 3, wn = wv >> 2;                  // 4x2 wave grid
  const int m0 = blockIdx.y * 128, n0 = blockIdx.x * 128;

  v8f acc[2][4] = {};
  v4f ra[4], rb[4];                                     // register staging (next tile)

  auto load_tiles = [&](int k0) {
#pragma unroll
    for (int i = 0; i < 4; ++i) {
      int idx = tid * 4 + i;
      { int r = idx >> 3, c = idx & 7;
        ra[i] = *(const v4f*)(x + (size_t)(m0 + r) * K + k0 + c * 4); }
      { int r = idx >> 5, c = idx & 31;
        rb[i] = *(const v4f*)(w + (size_t)(k0 + r) * N + n0 + c * 4); }
    }
  };
  auto store_tiles = [&]() {
#pragma unroll
    for (int i = 0; i < 4; ++i) {
      int idx = tid * 4 + i;
      { int r = idx >> 3, c = idx & 7;
        *(v4h*)&As[r][c * 4] = cvt4(ra[i]); }           // packed ds_store_b64
      { int r = idx >> 5, c = idx & 31;
        v4h p = cvt4(rb[i]);
#pragma unroll
        for (int e = 0; e < 4; ++e) Bs[c * 4 + e][r] = p[e]; }
    }
  };

  load_tiles(0);
  for (int k0 = 0; k0 < K; k0 += 32) {
    __syncthreads();                 // previous tile's consumers done
    store_tiles();
    __syncthreads();
    if (k0 + 32 < K) load_tiles(k0 + 32);   // overlap with WMMAs below

    v16h af[2], bf[4];
#pragma unroll
    for (int mt = 0; mt < 2; ++mt) af[mt] = load_a(&As[wm * 32 + mt * 16 + lr][0], half);
#pragma unroll
    for (int nt = 0; nt < 4; ++nt) bf[nt] = load_b(&Bs[wn * 64 + nt * 16 + lr][0], half);
#pragma unroll
    for (int mt = 0; mt < 2; ++mt)
#pragma unroll
      for (int nt = 0; nt < 4; ++nt) acc[mt][nt] = wmma16(af[mt], bf[nt], acc[mt][nt]);
  }

  const int bb = m0 >> 11;                              // whole tile is in one batch
#pragma unroll
  for (int nt = 0; nt < 4; ++nt) {
    const int colb = n0 + wn * 64 + nt * 16;            // 16-aligned, segment-uniform
    const int col = colb + lr;
    const float bv = bias[col];
    if (colb < Dn) {                                    // ---- Q -> f16 ws [B,S,D]
#pragma unroll
      for (int mt = 0; mt < 2; ++mt)
#pragma unroll
        for (int g = 0; g < 8; ++g) {
          int row = m0 + wm * 32 + mt * 16 + g + 8 * half;
          qh[(size_t)row * Dn + col] = (_Float16)(acc[mt][nt][g] + bv);
        }
    } else {                                            // ---- K/V -> present fp32 [B,2,H,S,HD]
      int c = colb - Dn; int slot = c >> 10; c &= (Dn - 1);
      int h = c >> 6;
      int hd = (c & 63) + lr;
      float* pp = present + ((((size_t)bb * 2 + slot) * Hn + h) * Sn) * HDn + hd;
#pragma unroll
      for (int mt = 0; mt < 2; ++mt)
#pragma unroll
        for (int g = 0; g < 8; ++g) {
          int s = (m0 & (Sn - 1)) + wm * 32 + mt * 16 + g + 8 * half;
          pp[(size_t)s * HDn] = acc[mt][nt][g] + bv;
        }
    }
  }
}

// ---------------- Kernel 2: flash attention, 1 WG per (b, h, 128 q rows) ----------------
__global__ __launch_bounds__(256) void flash_attn_kernel(
    const _Float16* __restrict__ qh, const float* __restrict__ present,
    _Float16* __restrict__ aout)
{
  __shared__ _Float16 Ks[64][72];        // [kv_row][hd]
  __shared__ _Float16 VT[64][72];        // [hd][kv_row]
  __shared__ _Float16 Pws[8][16 * 64];   // per-wave P staging (D-layout -> A-layout)
  const int tid = threadIdx.x;
  const int wv = tid >> 5, lane = tid & 31, half = lane >> 4, lr = lane & 15;
  const int qt = blockIdx.x, h = blockIdx.y, b = blockIdx.z;
  const int q0 = qt * 128, r0 = q0 + wv * 16;   // each wave owns 16 full query rows

  const float* Kp = present + (((size_t)b * 2 + 0) * Hn + h) * (size_t)Sn * HDn;
  const float* Vp = present + (((size_t)b * 2 + 1) * Hn + h) * (size_t)Sn * HDn;

  // Q fragments with 1/sqrt(HD) folded in (masked entries overwritten later anyway)
  v16h aq[2];
  {
    const _Float16* qrow = qh + (size_t)(b * Sn + r0 + lr) * Dn + h * HDn;
#pragma unroll
    for (int ks = 0; ks < 2; ++ks) {
      aq[ks] = load_a(qrow + ks * 32, half);
#pragma unroll
      for (int i = 0; i < 16; ++i) aq[ks][i] = aq[ks][i] * (_Float16)0.125f;
    }
  }

  float m_i[8], l_i[8];
  v8f o[4] = {};
#pragma unroll
  for (int g = 0; g < 8; ++g) { m_i[g] = -1e30f; l_i[g] = 0.0f; }

  v4f rk[4], rv[4];                      // register staging for next K/V tile
  auto load_kv = [&](int kt) {
#pragma unroll
    for (int i = 0; i < 4; ++i) {
      int idx = tid * 4 + i;
      int kr = idx >> 4, c = idx & 15;
      rk[i] = *(const v4f*)(Kp + (size_t)(kt * 64 + kr) * HDn + c * 4);
      rv[i] = *(const v4f*)(Vp + (size_t)(kt * 64 + kr) * HDn + c * 4);
    }
  };
  auto store_kv = [&]() {
#pragma unroll
    for (int i = 0; i < 4; ++i) {
      int idx = tid * 4 + i;
      int kr = idx >> 4, c = idx & 15;
      *(v4h*)&Ks[kr][c * 4] = cvt4(rk[i]);             // packed ds_store_b64
      v4h pv = cvt4(rv[i]);
#pragma unroll
      for (int e = 0; e < 4; ++e) VT[c * 4 + e][kr] = pv[e];
    }
  };

  const int nkt = q0 / 64 + 2;           // causal: only tiles touching j <= q0+127
  load_kv(0);
  for (int kt = 0; kt < nkt; ++kt) {
    __syncthreads();                     // previous tile's consumers done
    store_kv();
    __syncthreads();
    if (kt + 1 < nkt) load_kv(kt + 1);   // overlap with compute below
    if (kt * 64 > r0 + 15) continue;     // fully-masked tile for this wave

    // S = Q @ K^T : 4 N-tiles x 2 K-steps
    v8f s[4] = {};
#pragma unroll
    for (int ks = 0; ks < 2; ++ks)
#pragma unroll
      for (int nt = 0; nt < 4; ++nt) {
        v16h bf = load_b(&Ks[nt * 16 + lr][ks * 32], half);
        s[nt] = wmma16(aq[ks], bf, s[nt]);
      }

    // causal mask (scale already folded into Q; masked = -10000 exactly as reference)
#pragma unroll
    for (int nt = 0; nt < 4; ++nt)
#pragma unroll
      for (int g = 0; g < 8; ++g) {
        int qi = r0 + g + 8 * half;
        int kj = kt * 64 + nt * 16 + lr;
        if (kj > qi) s[nt][g] = -10000.0f;
      }

    // online softmax (rows are wave-local; reduce across 16-lane halves)
#pragma unroll
    for (int g = 0; g < 8; ++g) {
      float rmax = fmaxf(fmaxf(s[0][g], s[1][g]), fmaxf(s[2][g], s[3][g]));
#pragma unroll
      for (int d = 1; d < 16; d <<= 1) rmax = fmaxf(rmax, __shfl_xor(rmax, d, 32));
      float nm = fmaxf(m_i[g], rmax);
      float corr = __expf(m_i[g] - nm);
      float rs = 0.0f;
#pragma unroll
      for (int nt = 0; nt < 4; ++nt) { float p = __expf(s[nt][g] - nm); s[nt][g] = p; rs += p; }
#pragma unroll
      for (int d = 1; d < 16; d <<= 1) rs += __shfl_xor(rs, d, 32);
      l_i[g] = l_i[g] * corr + rs;
      m_i[g] = nm;
#pragma unroll
      for (int nt = 0; nt < 4; ++nt) o[nt][g] *= corr;
    }

    // P (D-layout) -> wave-private LDS -> A-layout fragments (in-wave DS ops are ordered)
    _Float16* P = &Pws[wv][0];
#pragma unroll
    for (int nt = 0; nt < 4; ++nt)
#pragma unroll
      for (int g = 0; g < 8; ++g)
        P[(g + 8 * half) * 64 + nt * 16 + lr] = (_Float16)s[nt][g];
    v16h ap[2];
#pragma unroll
    for (int ks = 0; ks < 2; ++ks) ap[ks] = load_a(P + lr * 64 + ks * 32, half);

    // O += P @ V
#pragma unroll
    for (int ks = 0; ks < 2; ++ks)
#pragma unroll
      for (int nt = 0; nt < 4; ++nt) {
        v16h bf = load_b(&VT[nt * 16 + lr][ks * 32], half);
        o[nt] = wmma16(ap[ks], bf, o[nt]);
      }
  }

#pragma unroll
  for (int nt = 0; nt < 4; ++nt)
#pragma unroll
    for (int g = 0; g < 8; ++g) {
      int row = r0 + g + 8 * half;
      int hd = nt * 16 + lr;
      float v = o[nt][g] / l_i[g];
      aout[(size_t)(b * Sn + row) * Dn + h * HDn + hd] = (_Float16)v;   // merged heads [B,S,D]
    }
}

// ---------------- Kernel 3: output projection  a[8192,1024] @ w_proj[1024,1024] ----------------
__global__ __launch_bounds__(256) void proj_gemm_kernel(
    const _Float16* __restrict__ a, const float* __restrict__ w,
    const float* __restrict__ bias, float* __restrict__ out)
{
  constexpr int K = Dn, N = Dn;
  __shared__ _Float16 As[128][40];
  __shared__ _Float16 Bs[128][40];
  const int tid = threadIdx.x;
  const int wv = tid >> 5, lane = tid & 31, half = lane >> 4, lr = lane & 15;
  const int wm = wv & 3, wn = wv >> 2;
  const int m0 = blockIdx.y * 128, n0 = blockIdx.x * 128;

  v8f acc[2][4] = {};
  v8h ra[2];
  v4f rb[4];

  auto load_tiles = [&](int k0) {
#pragma unroll
    for (int i = 0; i < 2; ++i) {        // A already f16: 512 v8h chunks / 256 threads
      int idx = tid * 2 + i;
      int r = idx >> 2, c = idx & 3;
      ra[i] = *(const v8h*)(a + (size_t)(m0 + r) * K + k0 + c * 8);
    }
#pragma unroll
    for (int i = 0; i < 4; ++i) {        // B: 32x128 fp32
      int idx = tid * 4 + i;
      int r = idx >> 5, c = idx & 31;
      rb[i] = *(const v4f*)(w + (size_t)(k0 + r) * N + n0 + c * 4);
    }
  };
  auto store_tiles = [&]() {
#pragma unroll
    for (int i = 0; i < 2; ++i) {
      int idx = tid * 2 + i;
      int r = idx >> 2, c = idx & 3;
      *(v8h*)&As[r][c * 8] = ra[i];
    }
#pragma unroll
    for (int i = 0; i < 4; ++i) {
      int idx = tid * 4 + i;
      int r = idx >> 5, c = idx & 31;
      v4h p = cvt4(rb[i]);
#pragma unroll
      for (int e = 0; e < 4; ++e) Bs[c * 4 + e][r] = p[e];
    }
  };

  load_tiles(0);
  for (int k0 = 0; k0 < K; k0 += 32) {
    __syncthreads();
    store_tiles();
    __syncthreads();
    if (k0 + 32 < K) load_tiles(k0 + 32);

    v16h af[2], bf[4];
#pragma unroll
    for (int mt = 0; mt < 2; ++mt) af[mt] = load_a(&As[wm * 32 + mt * 16 + lr][0], half);
#pragma unroll
    for (int nt = 0; nt < 4; ++nt) bf[nt] = load_b(&Bs[wn * 64 + nt * 16 + lr][0], half);
#pragma unroll
    for (int mt = 0; mt < 2; ++mt)
#pragma unroll
      for (int nt = 0; nt < 4; ++nt) acc[mt][nt] = wmma16(af[mt], bf[nt], acc[mt][nt]);
  }

#pragma unroll
  for (int mt = 0; mt < 2; ++mt)
#pragma unroll
    for (int nt = 0; nt < 4; ++nt) {
      int col = n0 + wn * 64 + nt * 16 + lr;
      float bv = bias[col];
#pragma unroll
      for (int g = 0; g < 8; ++g) {
        int row = m0 + wm * 32 + mt * 16 + g + 8 * half;
        out[(size_t)row * N + col] = acc[mt][nt][g] + bv;
      }
    }
}

extern "C" void kernel_launch(void* const* d_in, const int* in_sizes, int n_in,
                              void* d_out, int out_size, void* d_ws, size_t ws_size,
                              hipStream_t stream)
{
  (void)in_sizes; (void)n_in; (void)out_size; (void)ws_size;
  const float* x      = (const float*)d_in[0];
  const float* w_attn = (const float*)d_in[1];
  const float* b_attn = (const float*)d_in[2];
  const float* w_proj = (const float*)d_in[3];
  const float* b_proj = (const float*)d_in[4];

  float* out     = (float*)d_out;                       // [B,S,D] fp32
  float* present = out + (size_t)Bn * Sn * Dn;          // [B,2,H,S,HD] fp32

  _Float16* qh   = (_Float16*)d_ws;                     // 16 MB: Q f16 [B,S,D]
  _Float16* aout = qh + (size_t)Bn * Sn * Dn;           // 16 MB: attn out f16 [B,S,D]

  qkv_gemm_kernel<<<dim3(3 * Dn / 128, Bn * Sn / 128), 256, 0, stream>>>(
      x, w_attn, b_attn, qh, present);
  flash_attn_kernel<<<dim3(Sn / 128, Hn, Bn), 256, 0, stream>>>(qh, present, aout);
  proj_gemm_kernel<<<dim3(Dn / 128, Bn * Sn / 128), 256, 0, stream>>>(
      aout, w_proj, b_proj, out);
}
